// VarianceAdaptor_10316511445697
// MI455X (gfx1250) — compile-verified, matrix operand order checked by reading, not run on hardware
//
#include <hip/hip_runtime.h>
#include <hip/hip_bf16.h>

// ---------------------------------------------------------------------------
// FastSpeech2 variance adaptor for MI455X (gfx1250, wave32, WMMA).
// Convs run as WMMA f16 GEMMs (3 shifted taps), fused with ReLU + LayerNorm.
// Input tiles staged into LDS via the CDNA5 async global->LDS path
// (GLOBAL_LOAD_ASYNC_TO_LDS_B128, ASYNCcnt-tracked).
// ---------------------------------------------------------------------------

typedef __attribute__((ext_vector_type(16))) _Float16 v16h;
typedef __attribute__((ext_vector_type(8)))  float    v8f;
typedef __attribute__((ext_vector_type(4)))  float    f32x4;
typedef __attribute__((ext_vector_type(4)))  int      i32x4;

// AS-qualified pointer types for the async global->LDS builtin.
typedef __attribute__((address_space(1))) i32x4* gv4i_p;   // global
typedef __attribute__((address_space(3))) i32x4* lv4i_p;   // LDS

#define BB       32
#define T_TEXT   200
#define MAX_MEL  2000
#define DD       256
#define FF       256
#define LDSX     264          // padded f16 row stride for x tile
#define LDSS     260          // padded f32 row stride for LN staging

#if defined(__has_builtin)
#if __has_builtin(__builtin_amdgcn_global_load_async_to_lds_b128)
#define USE_ASYNC_LDS 1
#endif
#endif

union Frag16 { v16h h; f32x4 f[2]; };

// ---------------------------------------------------------------------------
// Repack conv weights (K=3, D=256, F=256) f32 -> f16 WMMA B-fragments.
// Fragment index: ((k*8 + c)*16 + ntile); 512 f16 per fragment, 16 per lane.
// Lane L holds column n = ntile*16 + (L&15); element j maps to
// kk = c*32 + (j<8 ? (L>>4)*8 + j : 16 + (L>>4)*8 + (j-8)).
// ---------------------------------------------------------------------------
__global__ __launch_bounds__(256)
void repack_w_kernel(const float* __restrict__ w, _Float16* __restrict__ wf) {
  int i = blockIdx.x * 256 + threadIdx.x;          // 0 .. 196607
  int e    = i & 511;
  int frag = i >> 9;
  int nt = frag & 15;
  int kc = frag >> 4;
  int c  = kc & 7;
  int k  = kc >> 3;
  int lane = e >> 4;
  int j    = e & 15;
  int halfb = lane >> 4;
  int d = c * 32 + (j < 8 ? halfb * 8 + j : 16 + halfb * 8 + (j - 8));
  int n = nt * 16 + (lane & 15);
  wf[i] = (_Float16)w[((size_t)k * DD + d) * FF + n];
}

// f32 -> f16 elementwise
__global__ __launch_bounds__(256)
void cvt_f16_kernel(const float* __restrict__ src, _Float16* __restrict__ dst, int n) {
  int i = blockIdx.x * 256 + threadIdx.x;
  if (i < n) dst[i] = (_Float16)src[i];
}

// ---------------------------------------------------------------------------
// Fused Conv1D(K=3, 256->256, SAME) + ReLU + LayerNorm, f16 in/out, f32 acc.
// Block = 256 threads = 8 wave32. Block computes 16 time rows x all 256 F.
// Wave w owns N-tiles {2w, 2w+1}. 48 v_wmma_f32_16x16x32_f16 per wave.
// ---------------------------------------------------------------------------
__global__ __launch_bounds__(256)
void conv_ln_kernel(const _Float16* __restrict__ xin,
                    const _Float16* __restrict__ wfrag,
                    const float* __restrict__ bias,
                    const float* __restrict__ gamma,
                    const float* __restrict__ beta,
                    _Float16* __restrict__ out,
                    int T) {
  __shared__ _Float16 xs[18 * LDSX];      // rows t0-1 .. t0+16
  __shared__ float    st[16 * LDSS];      // post-ReLU staging for LN

  const int ntilesM = (T + 15) >> 4;
  const int bIdx = blockIdx.x / ntilesM;
  const int t0   = (blockIdx.x % ntilesM) << 4;

  // Stage 18x256 f16 input tile into LDS (zero-padded at sequence edges).
  for (int i = threadIdx.x; i < 576; i += 256) {    // 18 rows * 32 chunks of 8 f16
    int r  = i >> 5;
    int c8 = (i & 31) * 8;
    int trow = t0 + r - 1;
#ifdef USE_ASYNC_LDS
    if (trow >= 0 && trow < T) {
      // Direct global -> LDS (no VGPR round trip); tracked by ASYNCcnt.
      __builtin_amdgcn_global_load_async_to_lds_b128(
          (gv4i_p)(xin + ((size_t)(bIdx * T + trow) * DD + c8)),
          (lv4i_p)(&xs[r * LDSX + c8]),
          0, 0);
    } else {
      *(uint4*)(&xs[r * LDSX + c8]) = make_uint4(0u, 0u, 0u, 0u);
    }
#else
    uint4 v = make_uint4(0u, 0u, 0u, 0u);
    if (trow >= 0 && trow < T)
      v = *(const uint4*)(xin + ((size_t)(bIdx * T + trow) * DD + c8));
    *(uint4*)(&xs[r * LDSX + c8]) = v;
#endif
  }
#ifdef USE_ASYNC_LDS
#if __has_builtin(__builtin_amdgcn_s_wait_asynccnt)
  __builtin_amdgcn_s_wait_asynccnt(0);
#else
  asm volatile("s_wait_asynccnt 0x0" ::: "memory");
#endif
#endif
  __syncthreads();

  const int wave = threadIdx.x >> 5;
  const int lane = threadIdx.x & 31;
  const int half = lane >> 4;
  const int n0 = wave * 32 + (lane & 15);
  const int n1 = n0 + 16;

  v8f acc0, acc1;
  {
    float b0 = bias[n0], b1 = bias[n1];
    for (int i = 0; i < 8; i++) { acc0[i] = b0; acc1[i] = b1; }
  }

  // GEMM: 3 conv taps x 8 K-chunks of 32.
  for (int k = 0; k < 3; k++) {
    const int arow = (lane & 15) + k;
    for (int c = 0; c < 8; c++) {
      Frag16 a;
      a.f[0] = *(const f32x4*)&xs[arow * LDSX + c * 32 + half * 8];
      a.f[1] = *(const f32x4*)&xs[arow * LDSX + c * 32 + 16 + half * 8];

      const size_t fbase = (size_t)((k * 8 + c) * 16) * 512;
      Frag16 b0f, b1f;
      const f32x4* q0 = (const f32x4*)&wfrag[fbase + (size_t)(wave * 2) * 512 + lane * 16];
      b0f.f[0] = q0[0]; b0f.f[1] = q0[1];
      const f32x4* q1 = (const f32x4*)&wfrag[fbase + (size_t)(wave * 2 + 1) * 512 + lane * 16];
      b1f.f[0] = q1[0]; b1f.f[1] = q1[1];

      acc0 = __builtin_amdgcn_wmma_f32_16x16x32_f16(false, a.h, false, b0f.h,
                                                    (short)0, acc0, false, false);
      acc1 = __builtin_amdgcn_wmma_f32_16x16x32_f16(false, a.h, false, b1f.h,
                                                    (short)0, acc1, false, false);
    }
  }

  // ReLU then stage to LDS for cross-wave LayerNorm.
  for (int v = 0; v < 8; v++) {
    int m = v + half * 8;
    float x0 = acc0[v] > 0.f ? acc0[v] : 0.f;
    float x1 = acc1[v] > 0.f ? acc1[v] : 0.f;
    st[m * LDSS + n0] = x0;
    st[m * LDSS + n1] = x1;
  }
  __syncthreads();

  // LayerNorm over F=256 per time row; wave handles rows {2w, 2w+1}.
  for (int rr = 0; rr < 2; rr++) {
    int m = wave * 2 + rr;
    float vals[8];
    float s = 0.f, s2 = 0.f;
    for (int j = 0; j < 8; j++) {
      float x = st[m * LDSS + lane + j * 32];
      vals[j] = x; s += x; s2 += x * x;
    }
    for (int off = 16; off >= 1; off >>= 1) {
      s  += __shfl_xor(s,  off, 32);
      s2 += __shfl_xor(s2, off, 32);
    }
    float mu  = s * (1.f / 256.f);
    float var = s2 * (1.f / 256.f) - mu * mu;
    float rs  = rsqrtf(var + 1e-5f);
    int t = t0 + m;
    if (t < T) {
      for (int j = 0; j < 8; j++) {
        int n = lane + j * 32;
        float y = (vals[j] - mu) * rs * gamma[n] + beta[n];
        out[((size_t)(bIdx * T + t)) * FF + n] = (_Float16)y;
      }
    }
  }
}

// ---------------------------------------------------------------------------
// Final Linear(256 -> 1) + mask. One wave32 per row.
// ---------------------------------------------------------------------------
__global__ __launch_bounds__(256)
void linear_mask_kernel(const _Float16* __restrict__ h,
                        const float* __restrict__ wl,
                        const float* __restrict__ bl,
                        const unsigned char* __restrict__ mask,
                        float* __restrict__ out, int rows) {
  int row  = blockIdx.x * 8 + (threadIdx.x >> 5);
  int lane = threadIdx.x & 31;
  if (row >= rows) return;
  float s = 0.f;
  for (int j = 0; j < 8; j++) {
    int f = lane + j * 32;
    s += (float)h[(size_t)row * FF + f] * wl[f];
  }
  for (int off = 16; off >= 1; off >>= 1) s += __shfl_xor(s, off, 32);
  if (lane == 0) {
    float o = s + bl[0];
    out[row] = mask[row] ? 0.f : o;
  }
}

// ---------------------------------------------------------------------------
// Length regulation phase 1: per-batch cumsum of durations + searchsorted idx.
// One block per batch element.
// ---------------------------------------------------------------------------
__global__ __launch_bounds__(256)
void lenreg_kernel(const long long* __restrict__ durations,
                   int* __restrict__ idxbuf,
                   float* __restrict__ mel_lens_out) {
  __shared__ long long cds[256];
  const int b = blockIdx.x;
  const int t = threadIdx.x;
  long long d = 0;
  if (t < T_TEXT) {
    long long x = durations[b * T_TEXT + t];
    d = x < 0 ? 0 : x;
  }
  cds[t] = d;
  __syncthreads();
  for (int off = 1; off < 256; off <<= 1) {
    long long add = (t >= off) ? cds[t - off] : 0;
    __syncthreads();
    cds[t] += add;
    __syncthreads();
  }
  long long mel_len = cds[T_TEXT - 1];
  if (t == 0) mel_lens_out[b] = (float)mel_len;

  long long lim = mel_len < MAX_MEL ? mel_len : MAX_MEL;
  for (int pos = t; pos < MAX_MEL; pos += 256) {
    // searchsorted(cum, pos, side='right')
    int lo = 0, hi = T_TEXT;
    while (lo < hi) {
      int mid = (lo + hi) >> 1;
      if (cds[mid] <= (long long)pos) lo = mid + 1; else hi = mid;
    }
    int idx = lo > T_TEXT - 1 ? T_TEXT - 1 : lo;
    idxbuf[b * MAX_MEL + pos] = ((long long)pos < lim) ? idx : -1;
  }
}

// Phase 2: gather enc rows into reg_out (f32 output) + reg16 (f16 working copy).
__global__ __launch_bounds__(256)
void gather_kernel(const float* __restrict__ enc,
                   const int* __restrict__ idxbuf,
                   float* __restrict__ regout,
                   _Float16* __restrict__ reg16) {
  int row  = blockIdx.x * 8 + (threadIdx.x >> 5);   // b*MAX_MEL + pos
  int lane = threadIdx.x & 31;
  int b   = row / MAX_MEL;
  int idx = idxbuf[row];
  for (int j = 0; j < 8; j++) {
    int dcol = lane + j * 32;
    float v = (idx >= 0) ? enc[((size_t)(b * T_TEXT + idx)) * DD + dcol] : 0.f;
    regout[(size_t)row * DD + dcol] = v;
    reg16[(size_t)row * DD + dcol]  = (_Float16)v;
  }
}

// ---------------------------------------------------------------------------
// Bucketize(target) via closed form on linspace(-3,3,255) + embed add.
// Updates fp32 reg_out (final output) and f16 working copy.
// ---------------------------------------------------------------------------
__global__ __launch_bounds__(256)
void embed_add_kernel(const float* __restrict__ target,
                      const float* __restrict__ embed,
                      float* __restrict__ regout,
                      _Float16* __restrict__ reg16) {
  int row  = blockIdx.x * 8 + (threadIdx.x >> 5);
  int lane = threadIdx.x & 31;
  float x = target[row];
  // idx = #{bins < x}, bins[i] = -3 + i*(6/254), i = 0..254  (side='left')
  float tt = (x + 3.0f) * (254.0f / 6.0f);
  int idx = (int)ceilf(tt);
  if (idx < 0) idx = 0;
  if (idx > 255) idx = 255;
  for (int j = 0; j < 8; j++) {
    int dcol = lane + j * 32;
    float nv = regout[(size_t)row * DD + dcol] + embed[(size_t)idx * DD + dcol];
    regout[(size_t)row * DD + dcol] = nv;
    reg16[(size_t)row * DD + dcol]  = (_Float16)nv;
  }
}

// ---------------------------------------------------------------------------
// Host-side orchestration.
// ---------------------------------------------------------------------------
extern "C" void kernel_launch(void* const* d_in, const int* in_sizes, int n_in,
                              void* d_out, int out_size, void* d_ws, size_t ws_size,
                              hipStream_t stream) {
  (void)in_sizes; (void)n_in; (void)out_size; (void)ws_size;

  const float*          enc_out    = (const float*)d_in[0];
  const unsigned char*  src_mask   = (const unsigned char*)d_in[1];
  const unsigned char*  mel_mask   = (const unsigned char*)d_in[2];
  const long long*      durations  = (const long long*)d_in[3];
  const float*          pitch_tgt  = (const float*)d_in[4];
  const float*          energy_tgt = (const float*)d_in[5];

  // params: dur (6..15), pitch (16..25), energy (26..35), embeds 36/37
  const float* P[30];
  for (int i = 0; i < 30; i++) P[i] = (const float*)d_in[6 + i];
  const float* pitch_embed  = (const float*)d_in[36];
  const float* energy_embed = (const float*)d_in[37];

  // d_out layout (floats): reg_out | mel_lens | dur_pred | pitch_pred | energy_pred
  float* out_reg    = (float*)d_out;
  float* out_mel    = out_reg + (size_t)BB * MAX_MEL * DD;
  float* out_dur    = out_mel + BB;
  float* out_pitch  = out_dur + (size_t)BB * T_TEXT;
  float* out_energy = out_pitch + (size_t)BB * MAX_MEL;

  // workspace layout
  char* p = (char*)d_ws;
  const size_t WFRAG_BYTES = (size_t)3 * 8 * 16 * 512 * sizeof(_Float16);  // 393216
  _Float16* wf[6];
  for (int i = 0; i < 6; i++) { wf[i] = (_Float16*)p; p += WFRAG_BYTES; }
  _Float16* enc16 = (_Float16*)p; p += (size_t)BB * T_TEXT * DD * sizeof(_Float16);
  _Float16* reg16 = (_Float16*)p; p += (size_t)BB * MAX_MEL * DD * sizeof(_Float16);
  _Float16* h1    = (_Float16*)p; p += (size_t)BB * MAX_MEL * DD * sizeof(_Float16);
  _Float16* h2    = (_Float16*)p; p += (size_t)BB * MAX_MEL * DD * sizeof(_Float16);
  int*      idxb  = (int*)p;      p += (size_t)BB * MAX_MEL * sizeof(int);

  // 1) repack conv weights: indices into P: w1=0, w2=4 within each predictor block of 10
  const int wsel[6] = {0, 4, 10, 14, 20, 24};
  for (int i = 0; i < 6; i++)
    repack_w_kernel<<<768, 256, 0, stream>>>(P[wsel[i]], wf[i]);

  // 2) enc_out -> f16
  {
    int n = BB * T_TEXT * DD;
    cvt_f16_kernel<<<(n + 255) / 256, 256, 0, stream>>>(enc_out, enc16, n);
  }

  const int gridDur = BB * ((T_TEXT + 15) / 16);   // 32 * 13
  const int gridMel = BB * (MAX_MEL / 16);         // 32 * 125
  const int rowsDur = BB * T_TEXT;
  const int rowsMel = BB * MAX_MEL;

  // 3) duration predictor
  conv_ln_kernel<<<gridDur, 256, 0, stream>>>(enc16, wf[0], P[1], P[2], P[3], h1, T_TEXT);
  conv_ln_kernel<<<gridDur, 256, 0, stream>>>(h1,    wf[1], P[5], P[6], P[7], h2, T_TEXT);
  linear_mask_kernel<<<rowsDur / 8, 256, 0, stream>>>(h2, P[8], P[9], src_mask, out_dur, rowsDur);

  // 4) length regulate
  lenreg_kernel<<<BB, 256, 0, stream>>>(durations, idxb, out_mel);
  gather_kernel<<<rowsMel / 8, 256, 0, stream>>>(enc_out, idxb, out_reg, reg16);

  // 5) pitch predictor
  conv_ln_kernel<<<gridMel, 256, 0, stream>>>(reg16, wf[2], P[11], P[12], P[13], h1, MAX_MEL);
  conv_ln_kernel<<<gridMel, 256, 0, stream>>>(h1,    wf[3], P[15], P[16], P[17], h2, MAX_MEL);
  linear_mask_kernel<<<rowsMel / 8, 256, 0, stream>>>(h2, P[18], P[19], mel_mask, out_pitch, rowsMel);

  // 6) add pitch embedding
  embed_add_kernel<<<rowsMel / 8, 256, 0, stream>>>(pitch_tgt, pitch_embed, out_reg, reg16);

  // 7) energy predictor
  conv_ln_kernel<<<gridMel, 256, 0, stream>>>(reg16, wf[4], P[21], P[22], P[23], h1, MAX_MEL);
  conv_ln_kernel<<<gridMel, 256, 0, stream>>>(h1,    wf[5], P[25], P[26], P[27], h2, MAX_MEL);
  linear_mask_kernel<<<rowsMel / 8, 256, 0, stream>>>(h2, P[28], P[29], mel_mask, out_energy, rowsMel);

  // 8) add energy embedding (final reg_out)
  embed_add_kernel<<<rowsMel / 8, 256, 0, stream>>>(energy_tgt, energy_embed, out_reg, reg16);
}
